// BertSelfAttention_63917703299369
// MI455X (gfx1250) — matrix operand order repaired and checked by reading, not compiled, not run on hardware
//
#include <hip/hip_runtime.h>
#include <hip/hip_bf16.h>
#include <math.h>

// ---------------------------------------------------------------------------
// BERT self-attention for MI455X (gfx1250, wave32, WMMA, async-to-LDS).
// Pipeline:
//   1) cast_x      : X fp32 -> bf16
//   2) cast_wT     : W[in,out] fp32 -> Wt[out,in] bf16
//   3) qkv_gemm    : bf16 WMMA 16x16x32, f32 accum, +bias; Q,K as [b,h,s,d],
//                    V transposed [b,h,d,s]
//   4) attention   : flash-style online softmax; K/V chunks double-buffered in
//                    LDS via GLOBAL_LOAD_ASYNC_TO_LDS_B128 (ASYNCcnt), 32
//                    queries per wave, WMMA for Q.K^T and P.V
// ---------------------------------------------------------------------------

typedef __bf16 bhalf;
typedef __attribute__((ext_vector_type(16))) __bf16 v16bf;
typedef __attribute__((ext_vector_type(8)))  __bf16 v8bf;
typedef __attribute__((ext_vector_type(4)))  __bf16 v4bf;
typedef __attribute__((ext_vector_type(8)))  float  v8f;

#define HIDDEN   1024
#define HEADS    16
#define HEAD_DIM 64
#define SEQ      2048
#define BATCH    2
#define ROWS     (BATCH * SEQ)     // 4096

// A-matrix 16x32 bf16 fragment (ISA 7.12.2): lane = row m, element e holds
// k = 8*hsel + 16*(e>>3) + (e&7)  -> two contiguous 16B loads.
__device__ __forceinline__ v16bf load_frag_a(const bhalf* __restrict__ p, int hsel) {
  v8bf lo = *(const v8bf*)(p + 8 * hsel);
  v8bf hi = *(const v8bf*)(p + 16 + 8 * hsel);
  v16bf r;
#pragma unroll
  for (int i = 0; i < 8; ++i) { r[i] = lo[i]; r[i + 8] = hi[i]; }
  return r;
}

// B-matrix 32x16 bf16 fragment: lane = column n, element e holds
// k = 8*hsel + 16*(e>>3) + (e&7) -> two 16B loads. p points at (row n, k_base).
__device__ __forceinline__ v16bf load_frag_b(const bhalf* __restrict__ p, int hsel) {
  v8bf lo = *(const v8bf*)(p + 8 * hsel);
  v8bf hi = *(const v8bf*)(p + 16 + 8 * hsel);
  v16bf r;
#pragma unroll
  for (int i = 0; i < 8; ++i) { r[i] = lo[i]; r[i + 8] = hi[i]; }
  return r;
}

__device__ __forceinline__ v8f wmma_bf16(v16bf a, v16bf b, v8f c) {
  return __builtin_amdgcn_wmma_f32_16x16x32_bf16(false, a, false, b, (short)0, c,
                                                 false, false);
}

// Async global->LDS copy, 16B per lane, tracked with ASYNCcnt (no VGPR data).
// lds_off = low 32 bits of generic shared pointer == workgroup LDS offset.
__device__ __forceinline__ void async_b128(unsigned lds_off, const bhalf* g) {
  asm volatile("global_load_async_to_lds_b128 %0, %1, off"
               :: "v"(lds_off), "v"(g) : "memory");
}
__device__ __forceinline__ void wait_async0() {
  asm volatile("s_wait_asynccnt 0x0" ::: "memory");
}

// ------------------------------ cast kernels -------------------------------

__global__ __launch_bounds__(256) void cast_x(const float* __restrict__ in,
                                              bhalf* __restrict__ out) {
  int i = (blockIdx.x * 256 + threadIdx.x) * 4;  // ROWS*HIDDEN elements, exact
  float4 v = *(const float4*)(in + i);
  v4bf o;
  o[0] = (bhalf)v.x; o[1] = (bhalf)v.y; o[2] = (bhalf)v.z; o[3] = (bhalf)v.w;
  *(v4bf*)(out + i) = o;
}

// W stored [in=k][out=n]; emit Wt[n][k] bf16 so GEMM B-frags are k-contiguous.
__global__ __launch_bounds__(256) void cast_wT(const float* __restrict__ w0,
                                               const float* __restrict__ w1,
                                               const float* __restrict__ w2,
                                               bhalf* __restrict__ out) {
  int z = blockIdx.z;
  const float* in = (z == 0) ? w0 : (z == 1) ? w1 : w2;
  bhalf* o = out + (size_t)z * HIDDEN * HIDDEN;
  int idx = blockIdx.x * 256 + threadIdx.x;      // 1M elements, read-coalesced
  int k = idx >> 10, n = idx & 1023;
  o[(size_t)n * HIDDEN + k] = (bhalf)in[idx];
}

// ------------------------------- QKV GEMM ----------------------------------
// C[4096,1024] = Xb @ Wt^T (+bias). Block tile 128x128, K-step 32, 8 waves:
// wave (wm,wn) in 2x4 -> 64x32 sub-tile -> 4x2 accumulators, 8 WMMAs / k-step.

__global__ __launch_bounds__(256)
void qkv_gemm(const bhalf* __restrict__ Xb, const bhalf* __restrict__ WtAll,
              const float* __restrict__ qb, const float* __restrict__ kb,
              const float* __restrict__ vb,
              bhalf* __restrict__ Qb, bhalf* __restrict__ Kb,
              bhalf* __restrict__ Vt) {
  __shared__ bhalf lA[128 * 32];
  __shared__ bhalf lB[128 * 32];

  const int z  = blockIdx.z;                       // 0=q, 1=k, 2=v
  const bhalf* Wt = WtAll + (size_t)z * HIDDEN * HIDDEN;
  const float* bias = (z == 0) ? qb : (z == 1) ? kb : vb;
  bhalf* outQK = (z == 0) ? Qb : Kb;

  const int m0 = blockIdx.y * 128;
  const int n0 = blockIdx.x * 128;
  const int tid = threadIdx.x;
  const int lane = tid & 31, wave = tid >> 5;
  const int hsel = lane >> 4, lm = lane & 15;
  const int wm = wave & 1, wn = wave >> 1;

  v8f acc[4][2] = {};

  const int srow = tid >> 1;
  const int scol = (tid & 1) * 16;

  for (int k0 = 0; k0 < HIDDEN; k0 += 32) {
    *(v8bf*)&lA[srow * 32 + scol]     = *(const v8bf*)&Xb[(size_t)(m0 + srow) * HIDDEN + k0 + scol];
    *(v8bf*)&lA[srow * 32 + scol + 8] = *(const v8bf*)&Xb[(size_t)(m0 + srow) * HIDDEN + k0 + scol + 8];
    *(v8bf*)&lB[srow * 32 + scol]     = *(const v8bf*)&Wt[(size_t)(n0 + srow) * HIDDEN + k0 + scol];
    *(v8bf*)&lB[srow * 32 + scol + 8] = *(const v8bf*)&Wt[(size_t)(n0 + srow) * HIDDEN + k0 + scol + 8];
    __syncthreads();

    v16bf af[4], bfr[2];
#pragma unroll
    for (int mi = 0; mi < 4; ++mi)
      af[mi] = load_frag_a(&lA[(wm * 64 + mi * 16 + lm) * 32], hsel);
#pragma unroll
    for (int ni = 0; ni < 2; ++ni)
      bfr[ni] = load_frag_b(&lB[(wn * 32 + ni * 16 + lm) * 32], hsel);

#pragma unroll
    for (int mi = 0; mi < 4; ++mi)
#pragma unroll
      for (int ni = 0; ni < 2; ++ni)
        acc[mi][ni] = wmma_bf16(af[mi], bfr[ni], acc[mi][ni]);
    __syncthreads();
  }

#pragma unroll
  for (int mi = 0; mi < 4; ++mi) {
#pragma unroll
    for (int ni = 0; ni < 2; ++ni) {
      const int col = n0 + wn * 32 + ni * 16 + lm;      // 0..1023
      const float bv = bias[col];
      const int h = col >> 6, d = col & 63;
#pragma unroll
      for (int r = 0; r < 8; ++r) {
        const int row = m0 + wm * 64 + mi * 16 + r + 8 * hsel;  // 0..4095
        const int b = row >> 11, s = row & (SEQ - 1);
        const bhalf v = (bhalf)(acc[mi][ni][r] + bv);
        if (z < 2)
          outQK[((size_t)(b * HEADS + h) * SEQ + s) * HEAD_DIM + d] = v;  // [b,h,s,d]
        else
          Vt[((size_t)(b * HEADS + h) * HEAD_DIM + d) * SEQ + s] = v;     // [b,h,d,s]
      }
    }
  }
}

// ------------------------------- Attention ---------------------------------
// grid (16 q-tiles, 32 bh). Block = 4 waves x 32 queries = 128 q rows.
// K/V 32-key chunks are double-buffered in LDS, staged with async-to-LDS
// (ASYNCcnt) while WMMAs consume the other buffer. Per chunk, per wave:
// 2 strips x (4 score WMMAs + online softmax + P->LDS->A-frag + 4 ctx WMMAs).

__global__ __launch_bounds__(128)
void attention(const bhalf* __restrict__ Qb, const bhalf* __restrict__ Kb,
               const bhalf* __restrict__ Vt, const float* __restrict__ mask,
               float* __restrict__ out) {
  __shared__ bhalf lK[2][32 * 64];     // [key][d]   4KB per buffer
  __shared__ bhalf lV[2][64 * 32];     // [d][key]   4KB per buffer
  __shared__ bhalf lP[4][2][16 * 32];  // per wave, per strip P scratch

  const int tid = threadIdx.x;
  const int lane = tid & 31, wave = tid >> 5;
  const int hsel = lane >> 4, lm = lane & 15;
  const int bh = blockIdx.y;
  const int b = bh >> 4, h = bh & 15;
  const int qbase = blockIdx.x * 128 + wave * 32;

  const bhalf* Qh = Qb + (size_t)bh * SEQ * HEAD_DIM;
  const bhalf* Kh = Kb + (size_t)bh * SEQ * HEAD_DIM;
  const bhalf* Vh = Vt + (size_t)bh * HEAD_DIM * SEQ;
  const float* mrow = mask + (size_t)b * SEQ;

  // Q fragments for both strips x both 32-wide d-slices, register-resident.
  v16bf aq[2][2];
#pragma unroll
  for (int st = 0; st < 2; ++st) {
    const bhalf* qrow = Qh + (size_t)(qbase + st * 16 + lm) * HEAD_DIM;
    aq[st][0] = load_frag_a(qrow, hsel);
    aq[st][1] = load_frag_a(qrow + 32, hsel);
  }

  float mrun[2][8], lrun[2][8];
#pragma unroll
  for (int st = 0; st < 2; ++st)
#pragma unroll
    for (int r = 0; r < 8; ++r) { mrun[st][r] = -1e30f; lrun[st][r] = 0.0f; }
  v8f acc[2][4] = {};

  // Cooperative staging map: 128 threads x 64B (2x16B of K + 2x16B of V).
  const int krow = tid >> 2, kcol = (tid & 3) * 16;   // 32 x 64 K chunk
  const int vrow = tid >> 1, vcol = (tid & 1) * 16;   // 64 x 32 V chunk

  auto stage = [&](int buf, int kc) {
    const bhalf* gk = Kh + (size_t)(kc + krow) * HEAD_DIM + kcol;
    unsigned lk = (unsigned)(size_t)&lK[buf][krow * 64 + kcol];
    async_b128(lk, gk);
    async_b128(lk + 16, gk + 8);
    const bhalf* gv = Vh + (size_t)vrow * SEQ + kc + vcol;
    unsigned lv = (unsigned)(size_t)&lV[buf][vrow * 32 + vcol];
    async_b128(lv, gv);
    async_b128(lv + 16, gv + 8);
  };

  stage(0, 0);  // prologue fill of buffer 0

  for (int kc = 0, it = 0; kc < SEQ; kc += 32, ++it) {
    const int cur = it & 1;
    wait_async0();        // my async stores of buffer `cur` are done
    __syncthreads();      // everyone's are done & prior reads of cur^1 retired
    if (kc + 32 < SEQ) stage(cur ^ 1, kc + 32);

    const float mv0 = mrow[kc + lm];
    const float mv1 = mrow[kc + 16 + lm];

#pragma unroll
    for (int st = 0; st < 2; ++st) {
      // ---- scores: S[16 x 32] in two D-tiles, K frags from LDS ----
      v8f s[2] = {};
#pragma unroll
      for (int nt = 0; nt < 2; ++nt) {
        const bhalf* kr = &lK[cur][(nt * 16 + lm) * 64];   // n = key
        v16bf bk0 = load_frag_b(kr, hsel);                 // d 0..31
        v16bf bk1 = load_frag_b(kr + 32, hsel);            // d 32..63
        s[nt] = wmma_bf16(aq[st][0], bk0, s[nt]);
        s[nt] = wmma_bf16(aq[st][1], bk1, s[nt]);
      }
#pragma unroll
      for (int r = 0; r < 8; ++r) {
        s[0][r] = s[0][r] * 0.125f + mv0;   // 1/sqrt(64) + additive mask
        s[1][r] = s[1][r] * 0.125f + mv1;
      }

      // ---- online softmax (rows across 16 lanes of each half-wave) ----
      float corr[8];
#pragma unroll
      for (int r = 0; r < 8; ++r) {
        float mx = fmaxf(s[0][r], s[1][r]);
#pragma unroll
        for (int off = 1; off < 16; off <<= 1) mx = fmaxf(mx, __shfl_xor(mx, off, 32));
        const float nm = fmaxf(mrun[st][r], mx);
        corr[r] = __expf(mrun[st][r] - nm);
        mrun[st][r] = nm;
      }
      bhalf* myP = &lP[wave][st][0];
#pragma unroll
      for (int r = 0; r < 8; ++r) {
        const float p0 = __expf(s[0][r] - mrun[st][r]);
        const float p1 = __expf(s[1][r] - mrun[st][r]);
        float rs = p0 + p1;
#pragma unroll
        for (int off = 1; off < 16; off <<= 1) rs += __shfl_xor(rs, off, 32);
        lrun[st][r] = lrun[st][r] * corr[r] + rs;
        myP[(r + 8 * hsel) * 32 + lm]      = (bhalf)p0;   // D-layout -> LDS
        myP[(r + 8 * hsel) * 32 + 16 + lm] = (bhalf)p1;
#pragma unroll
        for (int t = 0; t < 4; ++t) acc[st][t][r] *= corr[r];
      }

      // ---- P back as A-frag (same-wave DS ops are in-order) ----
      const v16bf pa = load_frag_a(&myP[lm * 32], hsel);

      // ---- O += P V : V frags from LDS [d][key] ----
#pragma unroll
      for (int t = 0; t < 4; ++t) {
        v16bf bv = load_frag_b(&lV[cur][(t * 16 + lm) * 32], hsel);
        acc[st][t] = wmma_bf16(pa, bv, acc[st][t]);
      }
    }
  }

  // ---- normalize and store: out[b, s, h*64 + d] fp32 ----
#pragma unroll
  for (int st = 0; st < 2; ++st)
#pragma unroll
    for (int t = 0; t < 4; ++t)
#pragma unroll
      for (int r = 0; r < 8; ++r) {
        const int row = qbase + st * 16 + r + 8 * hsel;
        out[((size_t)(b * SEQ + row)) * HIDDEN + h * HEAD_DIM + t * 16 + lm] =
            acc[st][t][r] / lrun[st][r];
      }
}

// ------------------------------- launcher ----------------------------------

extern "C" void kernel_launch(void* const* d_in, const int* in_sizes, int n_in,
                              void* d_out, int out_size, void* d_ws, size_t ws_size,
                              hipStream_t stream) {
  const float* X    = (const float*)d_in[0];
  const float* msk  = (const float*)d_in[1];
  const float* q_w  = (const float*)d_in[2];
  const float* q_b  = (const float*)d_in[3];
  const float* k_w  = (const float*)d_in[4];
  const float* k_b  = (const float*)d_in[5];
  const float* v_w  = (const float*)d_in[6];
  const float* v_b  = (const float*)d_in[7];
  float* out = (float*)d_out;

  // Workspace layout (bf16), ~38 MB total.
  char* w = (char*)d_ws;
  const size_t MB = 1024 * 1024;
  bhalf* Xb = (bhalf*)(w);                  // 8 MB : [4096,1024]
  bhalf* Wt = (bhalf*)(w + 8 * MB);         // 6 MB : 3 x [1024(out),1024(in)]
  bhalf* Qb = (bhalf*)(w + 14 * MB);        // 8 MB : [b,h,s,d]
  bhalf* Kb = (bhalf*)(w + 22 * MB);        // 8 MB : [b,h,s,d]
  bhalf* Vt = (bhalf*)(w + 30 * MB);        // 8 MB : [b,h,d,s]

  cast_x<<<dim3(ROWS * HIDDEN / (256 * 4)), 256, 0, stream>>>(X, Xb);
  cast_wT<<<dim3(HIDDEN * HIDDEN / 256, 1, 3), 256, 0, stream>>>(q_w, k_w, v_w, Wt);
  qkv_gemm<<<dim3(HIDDEN / 128, ROWS / 128, 3), 256, 0, stream>>>(
      Xb, Wt, q_b, k_b, v_b, Qb, Kb, Vt);
  attention<<<dim3(SEQ / 128, BATCH * HEADS), 128, 0, stream>>>(Qb, Kb, Vt, msk, out);
}
